// SimplerMultiHeadAttention_53566832116155
// MI455X (gfx1250) — compile-verified, matrix operand order checked by reading, not run on hardware
//
#include <hip/hip_runtime.h>
#include <hip/hip_bf16.h>

// ---------------------------------------------------------------------------
// Problem constants (from reference): B,L,DM,DS,H,DK,DV = 8,1024,1024,512,16,64,64
// ---------------------------------------------------------------------------
constexpr int cB  = 8;
constexpr int cL  = 1024;
constexpr int cDM = 1024;
constexpr int cDS = 512;
constexpr int cH  = 16;
constexpr int cDK = 64;
constexpr int cKC = cDM + cDS;   // 1536 concat width
constexpr int cHD = cH * cDK;    // 1024

typedef __bf16 bf16;
typedef bf16 v16bf __attribute__((ext_vector_type(16)));
typedef float v8f   __attribute__((ext_vector_type(8)));
typedef unsigned int u32x4 __attribute__((ext_vector_type(4)));

// ---------------------------------------------------------------------------
// WMMA helpers (CDNA5 wave32, 16x16x32 bf16 -> f32 accumulate)
//
// A fragment (16x32, bf16): lane l holds row M=l%16; two contiguous 8-element
// (16-byte) runs at k0 + (l<16?0:8) and the same +16.
// B fragment (32x16, bf16): identical pattern with N=l%16 rows of a
// row-major [N,K] matrix.
// C/D (16x16 f32): VGPR e, lane l -> (M = e + 8*(l/16), N = l%16).
// ---------------------------------------------------------------------------
__device__ __forceinline__ v16bf frag_ld_bf16(const bf16* p) {
    union { u32x4 u[2]; v16bf v; } r;
    r.u[0] = *(const u32x4*)p;
    r.u[1] = *(const u32x4*)(p + 16);
    return r.v;
}

__device__ __forceinline__ v16bf frag_ld_f32(const float* p) {
    v16bf v;
#pragma unroll
    for (int i = 0; i < 8; ++i) v[i] = (bf16)p[i];
#pragma unroll
    for (int i = 0; i < 8; ++i) v[i + 8] = (bf16)p[i + 16];
    return v;
}

__device__ __forceinline__ v8f wmma_bf16(v16bf a, v16bf b, v8f c) {
    return __builtin_amdgcn_wmma_f32_16x16x32_bf16(false, a, false, b, (short)0, c,
                                                   false, false);
}

// CDNA5 async copy: global -> LDS, 16 bytes per lane, tracked by ASYNCcnt.
__device__ __forceinline__ void async_g2l_b128(unsigned lds_off, const bf16* gaddr) {
    asm volatile("global_load_async_to_lds_b128 %0, %1, off"
                 :: "v"(lds_off), "v"(gaddr) : "memory");
}
__device__ __forceinline__ void wait_async0() {
    asm volatile("s_wait_asynccnt 0x0" ::: "memory");
}

// ---------------------------------------------------------------------------
// Elementwise prep kernels
// ---------------------------------------------------------------------------

// out[b,l, 0:DM] = x[b,l,:] ; out[b,l, DM:KC] = state[b,:]   (bf16)
__global__ void k_concat(const float* __restrict__ x, const float* __restrict__ state,
                         bf16* __restrict__ out) {
    long i = (long)blockIdx.x * blockDim.x + threadIdx.x;
    long n = (long)cB * cL * cKC;
    if (i >= n) return;
    int c = (int)(i % cKC);
    long r = i / cKC;           // b*L + l
    int b = (int)(r / cL);
    float v = (c < cDM) ? x[r * cDM + c] : state[b * cDS + (c - cDM)];
    out[i] = (bf16)v;
}

__global__ void k_f2bf(const float* __restrict__ in, bf16* __restrict__ out, long n) {
    long i = (long)blockIdx.x * blockDim.x + threadIdx.x;
    if (i < n) out[i] = (bf16)in[i];
}

// Fill Xo[:, DM:KC] with broadcast state (the attn-out part is filled by k_attnV)
__global__ void k_fill_state(const float* __restrict__ state, bf16* __restrict__ Xo) {
    long i = (long)blockIdx.x * blockDim.x + threadIdx.x;
    long n = (long)cB * cL * cDS;
    if (i >= n) return;
    int c = (int)(i % cDS);
    long r = i / cDS;
    int b = (int)(r / cL);
    Xo[r * cKC + cDM + c] = (bf16)state[b * cDS + c];
}

// vh [b,l,h*64+d] -> vhT [(b*H+h)*64+d, l]
__global__ void k_transpose_vh(const bf16* __restrict__ vh, bf16* __restrict__ vhT) {
    long i = (long)blockIdx.x * blockDim.x + threadIdx.x;
    long n = (long)cB * cL * cHD;
    if (i >= n) return;
    int c = (int)(i % cHD);
    long r = i / cHD;
    int b = (int)(r / cL);
    int l = (int)(r % cL);
    int h = c >> 6, d = c & 63;
    vhT[(((long)(b * cH + h)) * cDK + d) * cL + l] = vh[i];
}

// ---------------------------------------------------------------------------
// Generic WMMA GEMM:  Y[M,N] = A[M,K] * B[N,K]^T + bias
// One wave: 16-row M tile x 64-col N group (4 WMMA tiles), A-fragment reuse.
// ---------------------------------------------------------------------------
__global__ void k_gemm_bf16out(const bf16* __restrict__ A, const bf16* __restrict__ Bm,
                               const float* __restrict__ bias, bf16* __restrict__ out,
                               int M, int N, int K, int lda, int ldb, int ldo) {
    int wave = blockIdx.x * (blockDim.x >> 5) + (threadIdx.x >> 5);
    int lane = threadIdx.x & 31;
    int Mt = M >> 4, NtG = N >> 6;
    if (wave >= Mt * NtG) return;              // wave-uniform
    int mi = wave / NtG, ng = wave % NtG;

    const bf16* Abase = A + (long)(mi * 16 + (lane & 15)) * lda + ((lane >> 4) << 3);
    v8f z = {0.f, 0.f, 0.f, 0.f, 0.f, 0.f, 0.f, 0.f};
    v8f acc[4] = {z, z, z, z};
    for (int k0 = 0; k0 < K; k0 += 32) {
        __builtin_prefetch((const void*)(Abase + k0 + 64), 0, 1);
        v16bf a = frag_ld_bf16(Abase + k0);
#pragma unroll
        for (int j = 0; j < 4; ++j) {
            const bf16* Bb = Bm + (long)(ng * 64 + j * 16 + (lane & 15)) * ldb +
                             ((lane >> 4) << 3) + k0;
            __builtin_prefetch((const void*)(Bb + 64), 0, 1);
            acc[j] = wmma_bf16(a, frag_ld_bf16(Bb), acc[j]);
        }
    }
    int mrow = mi * 16 + ((lane >> 4) << 3);
    int n0 = ng * 64 + (lane & 15);
#pragma unroll
    for (int j = 0; j < 4; ++j) {
        int n = n0 + j * 16;
        float bb = bias[n];
#pragma unroll
        for (int e = 0; e < 8; ++e)
            out[(long)(mrow + e) * ldo + n] = (bf16)(acc[j][e] + bb);
    }
}

// Same GEMM but fp32 output with bias + residual (for the fc projection)
__global__ void k_gemm_f32res(const bf16* __restrict__ A, const bf16* __restrict__ Bm,
                              const float* __restrict__ bias, const float* __restrict__ res,
                              float* __restrict__ out,
                              int M, int N, int K, int lda, int ldb, int ldo) {
    int wave = blockIdx.x * (blockDim.x >> 5) + (threadIdx.x >> 5);
    int lane = threadIdx.x & 31;
    int Mt = M >> 4, NtG = N >> 6;
    if (wave >= Mt * NtG) return;
    int mi = wave / NtG, ng = wave % NtG;

    const bf16* Abase = A + (long)(mi * 16 + (lane & 15)) * lda + ((lane >> 4) << 3);
    v8f z = {0.f, 0.f, 0.f, 0.f, 0.f, 0.f, 0.f, 0.f};
    v8f acc[4] = {z, z, z, z};
    for (int k0 = 0; k0 < K; k0 += 32) {
        __builtin_prefetch((const void*)(Abase + k0 + 64), 0, 1);
        v16bf a = frag_ld_bf16(Abase + k0);
#pragma unroll
        for (int j = 0; j < 4; ++j) {
            const bf16* Bb = Bm + (long)(ng * 64 + j * 16 + (lane & 15)) * ldb +
                             ((lane >> 4) << 3) + k0;
            __builtin_prefetch((const void*)(Bb + 64), 0, 1);
            acc[j] = wmma_bf16(a, frag_ld_bf16(Bb), acc[j]);
        }
    }
    int mrow = mi * 16 + ((lane >> 4) << 3);
    int n0 = ng * 64 + (lane & 15);
#pragma unroll
    for (int j = 0; j < 4; ++j) {
        int n = n0 + j * 16;
        float bb = bias[n];
#pragma unroll
        for (int e = 0; e < 8; ++e) {
            long idx = (long)(mrow + e) * ldo + n;
            out[idx] = acc[j][e] + bb + res[idx];
        }
    }
}

// ---------------------------------------------------------------------------
// scores[h,b,q,k] = (qh[b,q,h,:] . kh[b,k,h,:]) / 8, masked to -1e9
// Written directly into the attn output region in [h*B+b, q, k] layout.
// ---------------------------------------------------------------------------
__global__ void k_scores(const bf16* __restrict__ qh, const bf16* __restrict__ kh,
                         const unsigned char* __restrict__ mask, float* __restrict__ attn) {
    int wave = blockIdx.x * (blockDim.x >> 5) + (threadIdx.x >> 5);
    int lane = threadIdx.x & 31;
    // 128 bh * 64 q-tiles * 16 k-groups = 131072 waves
    int bh = wave >> 10;
    int r = wave & 1023;
    int qt = r >> 4;
    int kg = r & 15;
    int b = bh >> 4, h = bh & 15;

    const bf16* Qb = qh + (long)b * cL * cHD + h * cDK;
    const bf16* Kb = kh + (long)b * cL * cHD + h * cDK;
    const bf16* Abase = Qb + (long)(qt * 16 + (lane & 15)) * cHD + ((lane >> 4) << 3);

    v8f z = {0.f, 0.f, 0.f, 0.f, 0.f, 0.f, 0.f, 0.f};
    v8f acc[4] = {z, z, z, z};
#pragma unroll
    for (int k0 = 0; k0 < cDK; k0 += 32) {
        v16bf a = frag_ld_bf16(Abase + k0);
#pragma unroll
        for (int j = 0; j < 4; ++j) {
            const bf16* Bb = Kb + (long)(kg * 64 + j * 16 + (lane & 15)) * cHD +
                             ((lane >> 4) << 3) + k0;
            acc[j] = wmma_bf16(a, frag_ld_bf16(Bb), acc[j]);
        }
    }
    const float scale = 0.125f;   // 1/sqrt(64)
    float* arow = attn + ((long)(h * cB + b)) * cL * cL;
    int q0 = qt * 16 + ((lane >> 4) << 3);
    int kc = kg * 64 + (lane & 15);
#pragma unroll
    for (int j = 0; j < 4; ++j) {
        int kk = kc + j * 16;
#pragma unroll
        for (int e = 0; e < 8; ++e) {
            int qq = q0 + e;
            float s = acc[j][e] * scale;
            if (mask[((long)b * cL + qq) * cL + kk]) s = -1e9f;
            arow[(long)qq * cL + kk] = s;
        }
    }
}

// In-place row softmax over 1024 columns; one 256-thread block per row.
__global__ void k_softmax(float* __restrict__ attn) {
    long row = blockIdx.x;
    float* p = attn + row * cL;
    __shared__ float red[256];
    int t = threadIdx.x;
    float v0 = p[t], v1 = p[t + 256], v2 = p[t + 512], v3 = p[t + 768];
    float m = fmaxf(fmaxf(v0, v1), fmaxf(v2, v3));
    red[t] = m; __syncthreads();
    for (int s = 128; s > 0; s >>= 1) {
        if (t < s) red[t] = fmaxf(red[t], red[t + s]);
        __syncthreads();
    }
    m = red[0]; __syncthreads();
    v0 = expf(v0 - m); v1 = expf(v1 - m); v2 = expf(v2 - m); v3 = expf(v3 - m);
    red[t] = v0 + v1 + v2 + v3; __syncthreads();
    for (int s = 128; s > 0; s >>= 1) {
        if (t < s) red[t] += red[t + s];
        __syncthreads();
    }
    float inv = 1.0f / red[0];
    p[t] = v0 * inv; p[t + 256] = v1 * inv; p[t + 512] = v2 * inv; p[t + 768] = v3 * inv;
}

// ---------------------------------------------------------------------------
// out_attn[b,q,h*64+d] = sum_k attn[h,b,q,k] * vhT[b,h,d,k]  -> Xo[:, 0:DM]
//
// CDNA5 path: the 64x32 bf16 B-tile (vhT slice) is staged into LDS with
// GLOBAL_LOAD_ASYNC_TO_LDS_B128 (ASYNCcnt), double-buffered, DMA overlapped
// with WMMA. 8 waves per block share each tile; wave 0 drives the DMA.
// LDS rows padded to 80B so the 16-lane ds_load_b128 fragment reads are
// bank-conflict free (r*20 mod 64 distinct for r in [0,16)).
// ---------------------------------------------------------------------------
__global__ void __launch_bounds__(256) k_attnV(const float* __restrict__ attn,
                                               const bf16* __restrict__ vhT,
                                               bf16* __restrict__ Xo) {
    __shared__ __align__(16) bf16 tile[2][64][40];   // 2 x 64 rows x 80B
    int w = threadIdx.x >> 5, lane = threadIdx.x & 31;
    // 128 bh * 8 m-groups = 1024 blocks; wave w handles m-tile mg*8 + w
    int bh = blockIdx.x >> 3, mg = blockIdx.x & 7;
    int b = bh >> 4, h = bh & 15;
    int mt = mg * 8 + w;

    const float* Ab = attn + ((long)(h * cB + b)) * cL * cL +
                      (long)(mt * 16 + (lane & 15)) * cL + ((lane >> 4) << 3);
    const bf16* Bbase = vhT + ((long)(b * cH + h)) * cDK * cL;
    unsigned ldsbase = (unsigned)(unsigned long long)(void*)&tile[0][0][0];

    // DMA one 64x32 bf16 tile (k0..k0+31 of all 64 d-rows) into buffer `buf`.
    // 256 16B chunks / 32 lanes = 8 async instructions.
    auto stage = [&](int buf, int k0) {
#pragma unroll
        for (int i = 0; i < 8; ++i) {
            int c = lane + i * 32;          // 0..255
            int d = c >> 2, ch = c & 3;
            unsigned lds = ldsbase + (unsigned)(buf * 5120 + d * 80 + ch * 16);
            async_g2l_b128(lds, Bbase + (long)d * cL + k0 + ch * 8);
        }
    };

    if (w == 0) { stage(0, 0); wait_async0(); }
    __syncthreads();

    v8f z = {0.f, 0.f, 0.f, 0.f, 0.f, 0.f, 0.f, 0.f};
    v8f acc[4] = {z, z, z, z};
    for (int ks = 0; ks < cL / 32; ++ks) {
        int cur = ks & 1;
        if (w == 0 && ks + 1 < cL / 32) stage(cur ^ 1, (ks + 1) * 32);  // overlap DMA
        v16bf a = frag_ld_f32(Ab + ks * 32);   // fp32 attn -> bf16 fragment
#pragma unroll
        for (int j = 0; j < 4; ++j) {
            const bf16* Lp = &tile[cur][j * 16 + (lane & 15)][(lane >> 4) << 3];
            acc[j] = wmma_bf16(a, frag_ld_bf16(Lp), acc[j]);
        }
        __syncthreads();                     // everyone done reading `cur`
        if (w == 0) wait_async0();           // next buffer landed in LDS
        __syncthreads();
    }

    int q0 = mt * 16 + ((lane >> 4) << 3);
#pragma unroll
    for (int j = 0; j < 4; ++j) {
        int d = j * 16 + (lane & 15);
#pragma unroll
        for (int e = 0; e < 8; ++e)
            Xo[((long)b * cL + q0 + e) * cKC + h * cDK + d] = (bf16)acc[j][e];
    }
}

// ---------------------------------------------------------------------------
// Generic row LayerNorm (width 1024 or 512), one 256-thread block per row
// ---------------------------------------------------------------------------
__global__ void k_layernorm(const float* __restrict__ in, float* __restrict__ out,
                            const float* __restrict__ g, const float* __restrict__ bt,
                            int width) {
    long row = blockIdx.x;
    const float* p = in + row * width;
    float* o = out + row * width;
    __shared__ float red[256];
    int t = threadIdx.x;
    float s = 0.f;
    for (int c = t; c < width; c += 256) s += p[c];
    red[t] = s; __syncthreads();
    for (int k = 128; k > 0; k >>= 1) { if (t < k) red[t] += red[t + k]; __syncthreads(); }
    float mean = red[0] / (float)width; __syncthreads();
    s = 0.f;
    for (int c = t; c < width; c += 256) { float d = p[c] - mean; s += d * d; }
    red[t] = s; __syncthreads();
    for (int k = 128; k > 0; k >>= 1) { if (t < k) red[t] += red[t + k]; __syncthreads(); }
    float inv = rsqrtf(red[0] / (float)width + 1e-5f);
    for (int c = t; c < width; c += 256)
        o[c] = (p[c] - mean) * inv * g[c] + bt[c];
}

// ---------------------------------------------------------------------------
// Tiny state-query path (negligible FLOPs: plain VALU)
// ---------------------------------------------------------------------------
__global__ void k_sq(const float* __restrict__ state, const float* __restrict__ Wsq,
                     const float* __restrict__ bsq, float* __restrict__ sq) {
    int i = blockIdx.x * blockDim.x + threadIdx.x;
    if (i >= cB * cHD) return;
    int b = i >> 10, n = i & 1023;
    const float* s = state + b * cDS;
    const float* w = Wsq + (long)n * cDS;
    float acc = bsq[n];
    for (int k = 0; k < cDS; ++k) acc += s[k] * w[k];
    sq[i] = acc;
}

__global__ void k_state_attn(const float* __restrict__ sq, const bf16* __restrict__ kh,
                             const bf16* __restrict__ vh, float* __restrict__ s_out) {
    int bh = blockIdx.x;
    int b = bh >> 4, h = bh & 15;
    __shared__ float p[cL];
    __shared__ float red[256];
    int t = threadIdx.x;
    const float* qv = sq + b * cHD + h * cDK;
    float sc[4];
#pragma unroll
    for (int i = 0; i < 4; ++i) {
        int k = t + i * 256;
        const bf16* kr = kh + ((long)b * cL + k) * cHD + h * cDK;
        float acc = 0.f;
        for (int d = 0; d < cDK; ++d) acc += qv[d] * (float)kr[d];
        sc[i] = acc * 0.125f;
    }
    float m = fmaxf(fmaxf(sc[0], sc[1]), fmaxf(sc[2], sc[3]));
    red[t] = m; __syncthreads();
    for (int s = 128; s > 0; s >>= 1) { if (t < s) red[t] = fmaxf(red[t], red[t + s]); __syncthreads(); }
    m = red[0]; __syncthreads();
    float sum = 0.f;
#pragma unroll
    for (int i = 0; i < 4; ++i) { sc[i] = expf(sc[i] - m); sum += sc[i]; }
    red[t] = sum; __syncthreads();
    for (int s = 128; s > 0; s >>= 1) { if (t < s) red[t] += red[t + s]; __syncthreads(); }
    float inv = 1.0f / red[0]; __syncthreads();
#pragma unroll
    for (int i = 0; i < 4; ++i) p[t + i * 256] = sc[i] * inv;
    __syncthreads();
    if (t < cDK) {
        float acc = 0.f;
        for (int k = 0; k < cL; ++k)
            acc += p[k] * (float)vh[((long)b * cL + k) * cHD + h * cDK + t];
        s_out[b * cHD + h * cDK + t] = acc;
    }
}

__global__ void k_su(const float* __restrict__ s_out, const float* __restrict__ state,
                     const float* __restrict__ Wfs, const float* __restrict__ bfs,
                     float* __restrict__ su_pre) {
    int i = blockIdx.x * blockDim.x + threadIdx.x;
    if (i >= cB * cDS) return;
    int b = i / cDS, n = i % cDS;
    const float* w = Wfs + (long)n * cKC;
    float acc = bfs[n];
    for (int k = 0; k < cHD; ++k) acc += s_out[b * cHD + k] * w[k];
    for (int k = 0; k < cDS; ++k) acc += state[b * cDS + k] * w[cHD + k];
    su_pre[i] = acc;
}

// ---------------------------------------------------------------------------
// Host launcher
// ---------------------------------------------------------------------------
extern "C" void kernel_launch(void* const* d_in, const int* in_sizes, int n_in,
                              void* d_out, int out_size, void* d_ws, size_t ws_size,
                              hipStream_t stream) {
    const float* q     = (const float*)d_in[0];
    const float* k     = (const float*)d_in[1];
    const float* v     = (const float*)d_in[2];
    const float* state = (const float*)d_in[3];
    const unsigned char* mask = (const unsigned char*)d_in[4];   // jnp bool (1 byte)
    const float* Wq  = (const float*)d_in[5];
    const float* bq  = (const float*)d_in[6];
    const float* Wk  = (const float*)d_in[7];
    const float* bk  = (const float*)d_in[8];
    const float* Wv  = (const float*)d_in[9];
    const float* bv  = (const float*)d_in[10];
    const float* Wsq = (const float*)d_in[11];
    const float* bsq = (const float*)d_in[12];
    const float* Wfc = (const float*)d_in[13];
    const float* bfc = (const float*)d_in[14];
    const float* Wfs = (const float*)d_in[15];
    const float* bfs = (const float*)d_in[16];
    const float* ln_g  = (const float*)d_in[17];
    const float* ln_b  = (const float*)d_in[18];
    const float* sln_g = (const float*)d_in[19];
    const float* sln_b = (const float*)d_in[20];

    float* out_p  = (float*)d_out;                                  // [B,L,DM]
    float* su_p   = out_p + (long)cB * cL * cDM;                    // [B,DS]
    float* attn_p = su_p + (long)cB * cDS;                          // [H*B,L,L]

    // ---- workspace carving (256B aligned) ----
    char* wp = (char*)d_ws;
    auto alloc = [&](size_t bytes) {
        void* p = wp;
        wp += (bytes + 255) & ~(size_t)255;
        return p;
    };
    const long ML = (long)cB * cL;                 // 8192 rows
    bf16* Xq   = (bf16*)alloc(ML * cKC * 2);
    bf16* Xk   = (bf16*)alloc(ML * cKC * 2);
    bf16* Xv   = (bf16*)alloc(ML * cKC * 2);
    bf16* Wq_b = (bf16*)alloc((long)cHD * cKC * 2);
    bf16* Wk_b = (bf16*)alloc((long)cHD * cKC * 2);
    bf16* Wv_b = (bf16*)alloc((long)cHD * cKC * 2);
    bf16* Wfc_b= (bf16*)alloc((long)cDM * cKC * 2);
    bf16* qh   = (bf16*)alloc(ML * cHD * 2);
    bf16* kh   = (bf16*)alloc(ML * cHD * 2);
    bf16* vh   = (bf16*)alloc(ML * cHD * 2);
    bf16* vhT  = (bf16*)alloc(ML * cHD * 2);
    bf16* Xo   = (bf16*)alloc(ML * cKC * 2);
    float* preln = (float*)alloc(ML * cDM * 4);
    float* sq    = (float*)alloc((long)cB * cHD * 4);
    float* s_out = (float*)alloc((long)cB * cHD * 4);
    float* su_pre= (float*)alloc((long)cB * cDS * 4);

    // ---- 1) bf16 concat inputs + weight casts ----
    {
        long n = ML * cKC;
        int g = (int)((n + 255) / 256);
        k_concat<<<g, 256, 0, stream>>>(q, state, Xq);
        k_concat<<<g, 256, 0, stream>>>(k, state, Xk);
        k_concat<<<g, 256, 0, stream>>>(v, state, Xv);
        long wn = (long)cHD * cKC;
        int gw = (int)((wn + 255) / 256);
        k_f2bf<<<gw, 256, 0, stream>>>(Wq, Wq_b, wn);
        k_f2bf<<<gw, 256, 0, stream>>>(Wk, Wk_b, wn);
        k_f2bf<<<gw, 256, 0, stream>>>(Wv, Wv_b, wn);
        k_f2bf<<<gw, 256, 0, stream>>>(Wfc, Wfc_b, (long)cDM * cKC);
    }

    // ---- 2) projections (WMMA): [8192,1536] x [1024,1536]^T ----
    {
        int waves = (8192 / 16) * (1024 / 64);          // 8192
        int blocks = waves / 4;                         // 128 threads = 4 waves
        k_gemm_bf16out<<<blocks, 128, 0, stream>>>(Xq, Wq_b, bq, qh, 8192, 1024, 1536, cKC, cKC, cHD);
        k_gemm_bf16out<<<blocks, 128, 0, stream>>>(Xk, Wk_b, bk, kh, 8192, 1024, 1536, cKC, cKC, cHD);
        k_gemm_bf16out<<<blocks, 128, 0, stream>>>(Xv, Wv_b, bv, vh, 8192, 1024, 1536, cKC, cKC, cHD);
    }
    k_transpose_vh<<<(int)((ML * cHD + 255) / 256), 256, 0, stream>>>(vh, vhT);
    k_sq<<<(cB * cHD + 255) / 256, 256, 0, stream>>>(state, Wsq, bsq, sq);

    // ---- 3) scores (WMMA) -> attn region, masked + scaled ----
    k_scores<<<131072 / 8, 256, 0, stream>>>(qh, kh, mask, attn_p);

    // ---- 4) softmax in place over k (134M elements) ----
    k_softmax<<<cH * cB * cL, 256, 0, stream>>>(attn_p);

    // ---- 5) attn @ V (WMMA + async global->LDS double-buffered B tiles) ----
    k_attnV<<<1024, 256, 0, stream>>>(attn_p, vhT, Xo);
    k_fill_state<<<(int)((ML * cDS + 255) / 256), 256, 0, stream>>>(state, Xo);

    // ---- 6) output projection (WMMA) + residual, then LayerNorm ----
    k_gemm_f32res<<<8192 / 4, 128, 0, stream>>>(Xo, Wfc_b, bfc, q, preln,
                                                8192, 1024, 1536, cKC, cKC, cDM);
    k_layernorm<<<8192, 256, 0, stream>>>(preln, out_p, ln_g, ln_b, cDM);

    // ---- 7) state path (tiny) ----
    k_state_attn<<<cB * cH, 256, 0, stream>>>(sq, kh, vh, s_out);
    k_su<<<(cB * cDS + 255) / 256, 256, 0, stream>>>(s_out, state, Wfs, bfs, su_pre);
    k_layernorm<<<cB, 256, 0, stream>>>(su_pre, su_p, sln_g, sln_b, cDS);

    (void)in_sizes; (void)n_in; (void)out_size; (void)ws_size;
}
